// FlashInferAttention_61220463837532
// MI455X (gfx1250) — matrix operand order repaired and checked by reading, not compiled, not run on hardware
//
#include <hip/hip_runtime.h>
#include <math.h>

typedef __attribute__((ext_vector_type(16))) _Float16 v16h;
typedef __attribute__((ext_vector_type(8)))  _Float16 v8h;
typedef __attribute__((ext_vector_type(2)))  __fp16   v2fp;   // return type of cvt_pkrtz
typedef __attribute__((ext_vector_type(8)))  float    v8f;
typedef __attribute__((ext_vector_type(4)))  float    v4f;

#define NUM_HEADS    16
#define NUM_KV_HEADS 8
#define HEAD_DIM     128
#define QSTRIDE      (NUM_HEADS * HEAD_DIM)     // 2048
#define KSTRIDE      (NUM_KV_HEADS * HEAD_DIM)  // 1024
#define BLOCK_M      32                          // q rows per block (2 row-tiles)
#define BLOCK_N      32                          // keys per inner block
// (1/sqrt(128)) * log2(e): softmax done in exp2 domain
#define SCALE_LOG2E  0.12751744416076986f

union V16 { v16h v; v8h h[2]; };
union H8  { v8h v; v2fp p[4]; };

// raw hardware exp2 / rcp (no denormal fixup sequences)
__device__ inline float fast_exp2(float x) { return __builtin_amdgcn_exp2f(x); }
__device__ inline float fast_rcp (float x) { return __builtin_amdgcn_rcpf(x); }

// pack 8 contiguous fp32 -> 8 fp16 (packed converts)
__device__ inline v8h pack8(v4f f0, v4f f1) {
    H8 r;
    r.p[0] = __builtin_amdgcn_cvt_pkrtz(f0[0], f0[1]);
    r.p[1] = __builtin_amdgcn_cvt_pkrtz(f0[2], f0[3]);
    r.p[2] = __builtin_amdgcn_cvt_pkrtz(f1[0], f1[1]);
    r.p[3] = __builtin_amdgcn_cvt_pkrtz(f1[2], f1[3]);
    return r.v;
}

__global__ __launch_bounds__(128)
void fa_gqa_kernel(const float* __restrict__ q, const float* __restrict__ k,
                   const float* __restrict__ v, const int* __restrict__ pos,
                   float* __restrict__ out, int T)
{
    // LDS: K row-major [key][dim] f16; V transposed [dim][key] f16; P per-wave.
    __shared__ __align__(16) _Float16 sK [BLOCK_N * HEAD_DIM];   // 8 KB
    __shared__ __align__(16) _Float16 sVt[HEAD_DIM * BLOCK_N];   // 8 KB
    __shared__ __align__(16) _Float16 sP [4][16 * BLOCK_N];      // 4 KB
    __shared__ int sRed[4];

    const int tid  = threadIdx.x;
    const int wave = tid >> 5;
    const int lane = tid & 31;
    const int hi   = lane >> 4;        // 0 = lanes 0-15, 1 = lanes 16-31
    const int ln   = lane & 15;
    const int m0   = blockIdx.x * BLOCK_M;
    const int kvh  = blockIdx.y;
    const int head = kvh * 2 + (wave & 1);      // 2 q-heads per kv-head
    const int m0w  = m0 + (wave >> 1) * 16;     // this wave's 16-row tile

    // ---- per-row segment start / causal bound (matches C-tile row map) ----
    int start_r[8], i_r[8];
#pragma unroll
    for (int r = 0; r < 8; ++r) {
        int i = m0w + r + 8 * hi;
        i_r[r] = i;
        start_r[r] = i - pos[i];
    }
    int kb0 = start_r[0];
#pragma unroll
    for (int r = 1; r < 8; ++r) kb0 = min(kb0, start_r[r]);
#pragma unroll
    for (int m = 1; m < 32; m <<= 1) kb0 = min(kb0, __shfl_xor(kb0, m, 32));
    if (lane == 0) sRed[wave] = kb0;
    __syncthreads();
    int kb0_blk = min(min(sRed[0], sRed[1]), min(sRed[2], sRed[3])) & ~(BLOCK_N - 1);

    // ---- Q tile (16 x 128) -> WMMA-A layout registers ----
    V16 aq[4];
    {
        const float* qrow = q + (size_t)(m0w + ln) * QSTRIDE + head * HEAD_DIM;
#pragma unroll
        for (int kc = 0; kc < 4; ++kc) {
            int d0 = kc * 32 + hi * 8;
#pragma unroll
            for (int half = 0; half < 2; ++half) {
                v4f f0 = *(const v4f*)(qrow + d0 + 16 * half);
                v4f f1 = *(const v4f*)(qrow + d0 + 16 * half + 4);
                aq[kc].h[half] = pack8(f0, f1);
            }
        }
    }

    v8f acc[8] = {};
    float m_r[8], l_r[8];
#pragma unroll
    for (int r = 0; r < 8; ++r) { m_r[r] = -1e30f; l_r[r] = 0.0f; }

    const int jb_last = m0 + BLOCK_M - 1;

    for (int jb = kb0_blk; jb <= jb_last; jb += BLOCK_N) {
        // ---- staging: 128 threads, each handles 1/4 of one key row ----
        {
            const int key = tid >> 2;
            const int dq  = (tid & 3) * 32;
            const int kt  = jb + key;
            if (kt < T) {
                const float* krow = k + (size_t)kt * KSTRIDE + kvh * HEAD_DIM + dq;
                const float* vrow = v + (size_t)kt * KSTRIDE + kvh * HEAD_DIM + dq;
#pragma unroll
                for (int c = 0; c < 4; ++c) {
                    v4f f0 = *(const v4f*)(krow + c * 8);
                    v4f f1 = *(const v4f*)(krow + c * 8 + 4);
                    *(v8h*)(sK + key * HEAD_DIM + dq + c * 8) = pack8(f0, f1);
                }
#pragma unroll
                for (int d4 = 0; d4 < 8; ++d4) {
                    v4f f = *(const v4f*)(vrow + d4 * 4);
#pragma unroll
                    for (int e = 0; e < 4; ++e)
                        sVt[(dq + d4 * 4 + e) * BLOCK_N + key] = (_Float16)f[e];
                }
            }
            // prefetch next key block (one row per thread)
            if (tid < 32) {
                int nt = jb + BLOCK_N + tid;
                if (nt < T) __builtin_prefetch(k + (size_t)nt * KSTRIDE + kvh * HEAD_DIM, 0, 1);
            } else if (tid < 64) {
                int nt = jb + tid;              // == jb + 32 + (tid - 32)
                if (nt < T) __builtin_prefetch(v + (size_t)nt * KSTRIDE + kvh * HEAD_DIM, 0, 1);
            }
        }
        __syncthreads();

        // ---- S = Q K^T: preload ALL 8 B tiles, then 8 back-to-back WMMAs ----
        V16 bk[2][4];
#pragma unroll
        for (int nc = 0; nc < 2; ++nc) {
            const _Float16* kptr = sK + (nc * 16 + ln) * HEAD_DIM + hi * 8;
#pragma unroll
            for (int kc = 0; kc < 4; ++kc) {
                bk[nc][kc].h[0] = *(const v8h*)(kptr + kc * 32);
                bk[nc][kc].h[1] = *(const v8h*)(kptr + kc * 32 + 16);
            }
        }
        v8f s[2];
#pragma unroll
        for (int nc = 0; nc < 2; ++nc) {
            v8f c = {};
#pragma unroll
            for (int kc = 0; kc < 4; ++kc)
                c = __builtin_amdgcn_wmma_f32_16x16x32_f16(false, aq[kc].v, false, bk[nc][kc].v,
                                                           (short)0, c, false, false);
            s[nc] = c;
        }

        // ---- online softmax (exp2 domain): cross-lane reduce for max only ----
        float alpha[8];
#pragma unroll
        for (int r = 0; r < 8; ++r) {
            float sv[2]; bool val[2];
            float sm = -1e30f;
#pragma unroll
            for (int nc = 0; nc < 2; ++nc) {
                int j = jb + nc * 16 + ln;
                sv[nc]  = s[nc][r] * SCALE_LOG2E;
                val[nc] = (j >= start_r[r]) && (j <= i_r[r]);
                sm = fmaxf(sm, val[nc] ? sv[nc] : -1e30f);
            }
#pragma unroll
            for (int mm = 1; mm < 16; mm <<= 1) sm = fmaxf(sm, __shfl_xor(sm, mm, 32));
            float mnew = fmaxf(m_r[r], sm);
            float a    = fast_exp2(m_r[r] - mnew);
            m_r[r]  = mnew;
            alpha[r] = a;
            float rs = 0.0f;
#pragma unroll
            for (int nc = 0; nc < 2; ++nc) {
                float p = val[nc] ? fast_exp2(sv[nc] - mnew) : 0.0f;
                rs += p;
                sP[wave][(r + 8 * hi) * BLOCK_N + nc * 16 + ln] = (_Float16)p;
            }
            l_r[r] = l_r[r] * a + rs;      // per-lane partial; reduced once at end
        }
#pragma unroll
        for (int n = 0; n < 8; ++n)
#pragma unroll
            for (int r = 0; r < 8; ++r)
                acc[n][r] *= alpha[r];

        // ---- O += P V: preload A and ALL 8 B tiles, then 8 WMMAs ----
        V16 ap;
        ap.h[0] = *(const v8h*)(sP[wave] + ln * BLOCK_N + hi * 8);
        ap.h[1] = *(const v8h*)(sP[wave] + ln * BLOCK_N + 16 + hi * 8);
        V16 bv[8];
#pragma unroll
        for (int n = 0; n < 8; ++n) {
            const _Float16* vptr = sVt + (n * 16 + ln) * BLOCK_N + hi * 8;
            bv[n].h[0] = *(const v8h*)(vptr);
            bv[n].h[1] = *(const v8h*)(vptr + 16);
        }
#pragma unroll
        for (int n = 0; n < 8; ++n)
            acc[n] = __builtin_amdgcn_wmma_f32_16x16x32_f16(false, ap.v, false, bv[n].v,
                                                            (short)0, acc[n], false, false);
        __syncthreads();   // protect K/V tiles before next overwrite
    }

    // ---- final row-sum reduction, normalize, store fp32 ----
#pragma unroll
    for (int r = 0; r < 8; ++r) {
        float rs = l_r[r];
#pragma unroll
        for (int mm = 1; mm < 16; mm <<= 1) rs += __shfl_xor(rs, mm, 32);
        float rl = fast_rcp(rs);
        float* orow = out + (size_t)i_r[r] * QSTRIDE + head * HEAD_DIM + ln;
#pragma unroll
        for (int n = 0; n < 8; ++n)
            orow[n * 16] = acc[n][r] * rl;
    }
}

extern "C" void kernel_launch(void* const* d_in, const int* in_sizes, int n_in,
                              void* d_out, int out_size, void* d_ws, size_t ws_size,
                              hipStream_t stream) {
    const float* q   = (const float*)d_in[0];
    const float* k   = (const float*)d_in[1];
    const float* v   = (const float*)d_in[2];
    const int*   pos = (const int*)d_in[3];
    float* out = (float*)d_out;

    const int T = in_sizes[0] / QSTRIDE;   // 4096
    dim3 grid(T / BLOCK_M, NUM_KV_HEADS);
    dim3 block(128);
    fa_gqa_kernel<<<grid, block, 0, stream>>>(q, k, v, pos, out, T);
}
